// QuantizedLinear_48730698940432
// MI455X (gfx1250) — compile-verified, matrix operand order checked by reading, not run on hardware
//
#include <hip/hip_runtime.h>
#include <stdint.h>

// Problem sizes (fixed by the reference)
#define M_TOTAL 8192     // B*S = 4*2048
#define K_TOTAL 4096
#define N_TOTAL 16384

// GEMM tiling
#define BM 64            // block tile M
#define BN 256           // block tile N
#define KB 64            // K bytes per WMMA step (16x16x64 iu8)
#define NKSTEPS (K_TOTAL / KB)
#define LDT 80           // LDS row stride in bytes (16B aligned, padded vs 64)

typedef __attribute__((ext_vector_type(8))) int v8i_t;

// Workspace layout (bytes)
//   [0,        32 MiB)   : A int8 (M x K)
//   [32 MiB,   +32 KiB)  : per-row x scales (float[M])
//   [33587200, +64 MiB)  : W int8 (N x K)
#define A8_OFF 0ull
#define XS_OFF 33554432ull
#define W8_OFF 33587200ull

// ---------------------------------------------------------------------------
// Kernel 1: per-row absmax quantization of x -> int8 + scale_x
// ---------------------------------------------------------------------------
__global__ __launch_bounds__(256) void quant_rows_kernel(
    const float* __restrict__ x, int8_t* __restrict__ a8,
    float* __restrict__ xscale) {
  const int row = blockIdx.x;
  const int tid = threadIdx.x;
  const float4* xr = (const float4*)(x + (size_t)row * K_TOTAL);

  float4 f[4];
  float mx = 0.f;
#pragma unroll
  for (int i = 0; i < 4; ++i) {
    f[i] = xr[tid * 4 + i];
    mx = fmaxf(mx, fmaxf(fmaxf(fabsf(f[i].x), fabsf(f[i].y)),
                         fmaxf(fabsf(f[i].z), fabsf(f[i].w))));
  }

  __shared__ float red[256];
  red[tid] = mx;
  __syncthreads();
  for (int s = 128; s > 0; s >>= 1) {
    if (tid < s) red[tid] = fmaxf(red[tid], red[tid + s]);
    __syncthreads();
  }
  const float maxv = red[0];
  const float scale = (maxv == 0.f) ? 1.f : (maxv * (1.f / 127.f));
  const float inv = 1.f / scale;
  if (tid == 0) xscale[row] = scale;

  int4 packed;
  int* p = &packed.x;
#pragma unroll
  for (int i = 0; i < 4; ++i) {
    const float* v = (const float*)&f[i];
    int q0 = (int)fminf(fmaxf(rintf(v[0] * inv), -127.f), 127.f);
    int q1 = (int)fminf(fmaxf(rintf(v[1] * inv), -127.f), 127.f);
    int q2 = (int)fminf(fmaxf(rintf(v[2] * inv), -127.f), 127.f);
    int q3 = (int)fminf(fmaxf(rintf(v[3] * inv), -127.f), 127.f);
    p[i] = (q0 & 255) | ((q1 & 255) << 8) | ((q2 & 255) << 16) | (q3 << 24);
  }
  ((int4*)(a8 + (size_t)row * K_TOTAL))[tid] = packed;
}

// ---------------------------------------------------------------------------
// Kernel 2: repack int32-held weights to dense int8 (268 MB -> 64 MB)
// ---------------------------------------------------------------------------
__global__ __launch_bounds__(256) void wconv_kernel(
    const int* __restrict__ w32, int8_t* __restrict__ w8) {
  const size_t idx = (size_t)blockIdx.x * 256 + threadIdx.x;
  int4 w = ((const int4*)w32)[idx];
  int packed = (w.x & 255) | ((w.y & 255) << 8) | ((w.z & 255) << 16) |
               ((w.w & 255) << 24);
  ((int*)w8)[idx] = packed;
}

// ---------------------------------------------------------------------------
// Kernel 3: int8 GEMM via V_WMMA_I32_16X16X64_IU8, double-buffered LDS,
// batched fragment loads, fused dequant epilogue.
// Block = 256 threads (8 wave32), 2x4 wave grid. Wave tile 32(M)x64(N).
// ---------------------------------------------------------------------------
__global__ __launch_bounds__(256) void gemm_kernel(
    const int8_t* __restrict__ a8, const int8_t* __restrict__ w8,
    const float* __restrict__ xscale, const float* __restrict__ wscale,
    const float* __restrict__ bias, float* __restrict__ out) {
  __shared__ __align__(16) int8_t ldsA[2][BM * LDT];   // 2 x 5 KB
  __shared__ __align__(16) int8_t ldsB[2][BN * LDT];   // 2 x 20 KB

  const int tid = threadIdx.x;
  const int lane = tid & 31;
  const int wave = tid >> 5;      // 0..7
  const int wm = wave >> 2;       // 0..1 -> M half (32 rows)
  const int wn = wave & 3;        // 0..3 -> N quarter (64 cols)
  const int bn0 = blockIdx.x * BN;
  const int bm0 = blockIdx.y * BM;

  // acc[s*2 + h]: N-subtile s (16 cols), M-subtile h (16 rows)
  v8i_t acc[8] = {};

  // 8-bit WMMA fragment addressing:
  // lanes 0-15 = rows 0-15 with K chunks {0-7,16-23,32-39,48-55};
  // lanes 16-31 = same rows, K chunks +8.
  const int frow = lane & 15;
  const int koff = (lane < 16) ? 0 : 8;
  const int sr = tid >> 2;        // staging row
  const int sc = (tid & 3) * 16;  // staging chunk byte offset

  // ---- stage K-step 0 into buffer 0 ----
  {
    int4 va = *(const int4*)(a8 + (size_t)(bm0 + sr) * K_TOTAL + sc);
    *(int4*)(&ldsA[0][sr * LDT + sc]) = va;
#pragma unroll
    for (int p = 0; p < 4; ++p) {
      const int r = sr + p * 64;
      int4 vb = *(const int4*)(w8 + (size_t)(bn0 + r) * K_TOTAL + sc);
      *(int4*)(&ldsB[0][r * LDT + sc]) = vb;
    }
  }
  __syncthreads();

  for (int kt = 0; kt < NKSTEPS; ++kt) {
    const int cur = kt & 1;

    // ---- stage K-step kt+1 into the other buffer (overlaps compute) ----
    if (kt + 1 < NKSTEPS) {
      const int k1 = (kt + 1) * KB;
      int4 va = *(const int4*)(a8 + (size_t)(bm0 + sr) * K_TOTAL + k1 + sc);
      *(int4*)(&ldsA[cur ^ 1][sr * LDT + sc]) = va;
#pragma unroll
      for (int p = 0; p < 4; ++p) {
        const int r = sr + p * 64;
        int4 vb = *(const int4*)(w8 + (size_t)(bn0 + r) * K_TOTAL + k1 + sc);
        *(int4*)(&ldsB[cur ^ 1][r * LDT + sc]) = vb;
      }
      if (kt + 2 < NKSTEPS) {
        __builtin_prefetch(w8 + (size_t)(bn0 + sr) * K_TOTAL + k1 + KB, 0, 3);
      }
    }

    // ---- batch-load ALL fragments, then issue the WMMA chain ----
    union F { v8i_t v; uint64_t q[4]; };
    F af[2];
#pragma unroll
    for (int h = 0; h < 2; ++h) {
      const int8_t* base = &ldsA[cur][(wm * 32 + h * 16 + frow) * LDT + koff];
#pragma unroll
      for (int i = 0; i < 4; ++i) af[h].q[i] = *(const uint64_t*)(base + 16 * i);
    }
    F bf[4];
#pragma unroll
    for (int s = 0; s < 4; ++s) {
      const int8_t* base = &ldsB[cur][(wn * 64 + s * 16 + frow) * LDT + koff];
#pragma unroll
      for (int i = 0; i < 4; ++i) bf[s].q[i] = *(const uint64_t*)(base + 16 * i);
    }
#pragma unroll
    for (int s = 0; s < 4; ++s) {
      acc[s * 2 + 0] = __builtin_amdgcn_wmma_i32_16x16x64_iu8(
          true, af[0].v, true, bf[s].v, acc[s * 2 + 0], false, false);
      acc[s * 2 + 1] = __builtin_amdgcn_wmma_i32_16x16x64_iu8(
          true, af[1].v, true, bf[s].v, acc[s * 2 + 1], false, false);
    }
    __syncthreads();
  }

  // ---- epilogue: C/D layout: lane l, VGPR j -> m = j + 8*(l>=16), n = l&15
  const int n_in = lane & 15;
  const int mhalf = (lane >> 4) * 8;
  // per-lane row scales (16 rows this wave can touch)
  float xs_lo[8], xs_hi[8];
#pragma unroll
  for (int j = 0; j < 8; ++j) {
    xs_lo[j] = xscale[bm0 + wm * 32 + 0 + mhalf + j];
    xs_hi[j] = xscale[bm0 + wm * 32 + 16 + mhalf + j];
  }
#pragma unroll
  for (int s = 0; s < 4; ++s) {
    const int gn = bn0 + wn * 64 + s * 16 + n_in;
    const float wsc = wscale[gn];
    const float bs = bias[gn];
#pragma unroll
    for (int j = 0; j < 8; ++j) {
      const int gm0 = bm0 + wm * 32 + 0 + mhalf + j;
      const int gm1 = bm0 + wm * 32 + 16 + mhalf + j;
      out[(size_t)gm0 * N_TOTAL + gn] =
          (float)acc[s * 2 + 0][j] * xs_lo[j] * wsc + bs;
      out[(size_t)gm1 * N_TOTAL + gn] =
          (float)acc[s * 2 + 1][j] * xs_hi[j] * wsc + bs;
    }
  }
}

// ---------------------------------------------------------------------------
extern "C" void kernel_launch(void* const* d_in, const int* in_sizes, int n_in,
                              void* d_out, int out_size, void* d_ws,
                              size_t ws_size, hipStream_t stream) {
  (void)in_sizes; (void)n_in; (void)out_size; (void)ws_size;
  const float* x = (const float*)d_in[0];
  const int* w32 = (const int*)d_in[1];
  const float* wscale = (const float*)d_in[2];   // [N,1]
  const float* bias = (const float*)d_in[3];     // [N]
  float* out = (float*)d_out;

  int8_t* a8 = (int8_t*)d_ws + A8_OFF;
  float* xs = (float*)((char*)d_ws + XS_OFF);
  int8_t* w8 = (int8_t*)d_ws + W8_OFF;

  // 1) dynamic per-row quantization of x
  quant_rows_kernel<<<M_TOTAL, 256, 0, stream>>>(x, a8, xs);
  // 2) weight repack int32 -> int8 (N*K/4 ints)
  wconv_kernel<<<(N_TOTAL * (size_t)K_TOTAL) / 4 / 256, 256, 0, stream>>>(w32, w8);
  // 3) int8 WMMA GEMM + fused dequant/bias
  dim3 grid(N_TOTAL / BN, M_TOTAL / BM);   // 64 x 128
  gemm_kernel<<<grid, 256, 0, stream>>>(a8, w8, xs, wscale, bias, out);
}
